// DispersiveLoss_29961691856903
// MI455X (gfx1250) — matrix operand order repaired and checked by reading, not compiled, not run on hardware
//
#include <hip/hip_runtime.h>

typedef __attribute__((ext_vector_type(16))) _Float16 v16h;
typedef __attribute__((ext_vector_type(8)))  _Float16 v8h;
typedef __attribute__((ext_vector_type(8)))  float    v8f;
typedef __attribute__((ext_vector_type(4)))  float    v4f;

#define BB 8
#define SS 2048
#define DD 512
#define INV_T  10.0f     // 1 / 0.1
#define SIMMAX 10.0f     // sim = cos/T <= 1/T = 10 : fixed log-sum-exp shift
#define M_TILE 64        // rows per block (4 waves x 16)
#define N_TILE 64        // columns per LDS tile
#define KCH (DD / 32)    // 16 k-chunks of 32 for wmma f16
#define NTILES (SS / N_TILE)

// ---------------------------------------------------------------------------
// Kernel 1: L2-normalize rows (F.normalize semantics), emit f16 matrix.
// ---------------------------------------------------------------------------
__global__ __launch_bounds__(256) void norm_kernel(const float* __restrict__ x,
                                                   _Float16* __restrict__ ln) {
    const int row  = blockIdx.x * 8 + (threadIdx.x >> 5);
    const int lane = threadIdx.x & 31;
    const float* p = x + (size_t)row * DD + lane * 16;
    v4f a0 = *(const v4f*)(p + 0);
    v4f a1 = *(const v4f*)(p + 4);
    v4f a2 = *(const v4f*)(p + 8);
    v4f a3 = *(const v4f*)(p + 12);
    float ss = 0.f;
#pragma unroll
    for (int i = 0; i < 4; ++i)
        ss += a0[i]*a0[i] + a1[i]*a1[i] + a2[i]*a2[i] + a3[i]*a3[i];
#pragma unroll
    for (int off = 16; off; off >>= 1) ss += __shfl_xor(ss, off, 32);
    const float scale = 1.0f / fmaxf(sqrtf(ss), 1e-8f);

    v8h h0, h1;
#pragma unroll
    for (int i = 0; i < 4; ++i) {
        h0[i]     = (_Float16)(a0[i] * scale);
        h0[i + 4] = (_Float16)(a1[i] * scale);
        h1[i]     = (_Float16)(a2[i] * scale);
        h1[i + 4] = (_Float16)(a3[i] * scale);
    }
    _Float16* q = ln + (size_t)row * DD + lane * 16;
    *(v8h*)(q)     = h0;
    *(v8h*)(q + 8) = h1;
}

// ---------------------------------------------------------------------------
// Kernel 2: streaming Gram matrix + fixed-shift log-sum-exp.
// LDS column tiles in K-major chunk layout: phys(r,c) = c*1024B + r*16B.
//  -> B-fragment ds_load_b128: bank-conflict-free, all offsets are 16-bit
//     DS immediates (one address VGPR per buffer).
//  -> double-buffered 2x64KB; next-tile global loads overlap WMMAs.
//  -> TWO subtiles (E/O) interleaved per kc iteration: 2 independent WMMA
//     chains + 4 ds loads in flight -> LDS latency covered by 2 WMMAs.
// A operand (16 rows x K=512) resident in 128 VGPRs per wave.
// Fixed shift exp(sim-10): no cross-lane ops in the hot loop.
// ---------------------------------------------------------------------------
__global__ __launch_bounds__(128) void disp_kernel(const _Float16* __restrict__ ln,
                                                   float* __restrict__ partials) {
    __shared__ _Float16 smem[2][N_TILE * DD];   // 2 x 64 KB, K-major chunks

    const int tid  = threadIdx.x;
    const int lane = tid & 31;
    const int wave = tid >> 5;
    const int l15  = lane & 15;
    const int hi   = lane >> 4;              // 0: lanes 0-15, 1: lanes 16-31

    const int tilesPerBatch = SS / M_TILE;   // 32
    const int b     = blockIdx.x / tilesPerBatch;
    const int mTile = blockIdx.x % tilesPerBatch;
    const int mBase = mTile * M_TILE + wave * 16;
    const _Float16* base = ln + (size_t)b * SS * DD;

    // ---- preload A fragments (contiguous k = 16*hi + s per lane half), K=512 in 128 VGPRs
    v16h afrag[KCH];
    {
        const _Float16* arow = base + (size_t)(mBase + l15) * DD;
#pragma unroll
        for (int kc = 0; kc < KCH; ++kc) {
            const int k0 = kc * 32 + hi * 16;
            v8h lo = *(const v8h*)(arow + k0);
            v8h hh = *(const v8h*)(arow + k0 + 8);
            afrag[kc] = __builtin_shufflevector(lo, hh,
                0,1,2,3,4,5,6,7,8,9,10,11,12,13,14,15);
        }
    }

    // lane-private partial sums for this lane's 8 rows (rows mBase + hi*8 + j)
    float l[8], s[8];
#pragma unroll
    for (int j = 0; j < 8; ++j) { l[j] = 0.f; s[j] = 0.f; }

    // ---- staging bases (all loop-varying parts become immediates) ----
    const int sr = tid & 63;                  // tensor row this thread stages
    const int sc = tid >> 6;                  // chunk base (0 or 1)
    const _Float16* gstage = base + (size_t)sr * DD + sc * 8;
    _Float16* st0 = &smem[0][0] + sc * 512 + sr * 8;
    _Float16* st1 = &smem[1][0] + sc * 512 + sr * 8;
    const _Float16* bb0 = &smem[0][0] + l15 * 8 + hi * 1024;
    const _Float16* bb1 = &smem[1][0] + l15 * 8 + hi * 1024;

    // fixed-shift accumulation for one 16x16 subtile (wave-uniform diag check)
    auto update = [&](const v8f& c8, int colb) {
        if (colb == mBase) {                 // diagonal subtile
#pragma unroll
            for (int j = 0; j < 8; ++j) {
                const float v = c8[j] * INV_T;
                const bool diag = (l15 == hi * 8 + j);
                l[j] += diag ? 0.f : __expf(v - SIMMAX);
                s[j] += diag ? 0.f : v;
            }
        } else {
#pragma unroll
            for (int j = 0; j < 8; ++j) {
                const float v = c8[j] * INV_T;
                l[j] += __expf(v - SIMMAX);
                s[j] += v;
            }
        }
    };

    // prologue: stage tile 0 into buffer 0
    {
        const _Float16* g = gstage;           // colBase = 0
#pragma unroll
        for (int i = 0; i < 32; ++i) {
            v4f d = *(const v4f*)(g + i * 16);            // chunk c = sc + 2i
            *(v4f*)(st0 + i * 1024) = d;                  // phys: c*512h + r*8h
        }
    }
    __syncthreads();

#pragma unroll 1
    for (int ct = 0; ct < NTILES; ++ct) {
        const int colBase = ct * N_TILE;
        const _Float16* bb = (ct & 1) ? bb1 : bb0;
        _Float16*       st = (ct & 1) ? st0 : st1;
        const _Float16* gn = gstage + (size_t)(colBase + N_TILE) * DD;
        const bool havenext = (ct + 1 < NTILES);

#pragma unroll
        for (int p = 0; p < 2; ++p) {        // subtile pairs (2p, 2p+1)
            // issue next-tile global loads; WMMAs below cover their latency
            v4f reg[8];
            if (havenext) {
#pragma unroll
                for (int i = 0; i < 8; ++i)
                    reg[i] = *(const v4f*)(gn + ((2 * p) * 8 + i) * 16);
            }

            // ---- two interleaved 16x16 subtiles: 32 WMMAs over K=512 ----
            const int offE = p * 256;        // halves: subtile 2p
            const int offO = p * 256 + 128;  // halves: subtile 2p+1
            v8f cE = {0.f,0.f,0.f,0.f,0.f,0.f,0.f,0.f};
            v8f cO = {0.f,0.f,0.f,0.f,0.f,0.f,0.f,0.f};
            v8h eL = *(const v8h*)(bb + offE);
            v8h eH = *(const v8h*)(bb + offE + 512);
            v8h oL = *(const v8h*)(bb + offO);
            v8h oH = *(const v8h*)(bb + offO + 512);
#pragma unroll
            for (int kc = 0; kc < KCH; ++kc) {
                v8h eLn, eHn, oLn, oHn;
                if (kc + 1 < KCH) {                       // prefetch kc+1
                    const int o = (kc + 1) * 2048;
                    eLn = *(const v8h*)(bb + o + offE);
                    eHn = *(const v8h*)(bb + o + offE + 512);
                    oLn = *(const v8h*)(bb + o + offO);
                    oHn = *(const v8h*)(bb + o + offO + 512);
                }
                v16h bE = __builtin_shufflevector(eL, eH,
                    0,1,2,3,4,5,6,7,8,9,10,11,12,13,14,15);
                v16h bO = __builtin_shufflevector(oL, oH,
                    0,1,2,3,4,5,6,7,8,9,10,11,12,13,14,15);
                cE = __builtin_amdgcn_wmma_f32_16x16x32_f16(
                        false, afrag[kc], false, bE, (short)0, cE, false, false);
                cO = __builtin_amdgcn_wmma_f32_16x16x32_f16(
                        false, afrag[kc], false, bO, (short)0, cO, false, false);
                eL = eLn; eH = eHn; oL = oLn; oH = oHn;
            }

            if (havenext) {                  // store staging group 2p
#pragma unroll
                for (int i = 0; i < 8; ++i)
                    *(v4f*)(st + ((2 * p) * 8 + i) * 1024) = reg[i];
            }

            update(cE, colBase + (2 * p) * 16);

            // second staging group for this pair; updates cover load latency
            v4f reg2[8];
            if (havenext) {
#pragma unroll
                for (int i = 0; i < 8; ++i)
                    reg2[i] = *(const v4f*)(gn + ((2 * p + 1) * 8 + i) * 16);
            }

            update(cO, colBase + (2 * p + 1) * 16);

            if (havenext) {
#pragma unroll
                for (int i = 0; i < 8; ++i)
                    *(v4f*)(st + ((2 * p + 1) * 8 + i) * 1024) = reg2[i];
            }
        }
        __syncthreads();
    }

    // ---- one-time cross-lane reduction over the 16-lane N groups ----
#pragma unroll
    for (int j = 0; j < 8; ++j) {
#pragma unroll
        for (int off = 8; off; off >>= 1) {
            l[j] += __shfl_xor(l[j], off, 32);
            s[j] += __shfl_xor(s[j], off, 32);
        }
    }

    // per-row loss: (S-1)*(10 + log l) - s ; one partial per (wave, half)
    if (l15 == 0) {
        float csum = 0.f;
#pragma unroll
        for (int j = 0; j < 8; ++j)
            csum += (float)(SS - 1) * (SIMMAX + __logf(l[j])) - s[j];
        partials[blockIdx.x * 8 + wave * 2 + hi] = csum;
    }
}

// ---------------------------------------------------------------------------
// Kernel 3: deterministic fixed-order reduction of 2048 partials.
// ---------------------------------------------------------------------------
__global__ __launch_bounds__(256) void fin_kernel(const float* __restrict__ partials,
                                                  float* __restrict__ out) {
    __shared__ float red[256];
    float ssum = 0.f;
    for (int i = threadIdx.x; i < 2048; i += 256) ssum += partials[i];
    red[threadIdx.x] = ssum;
    __syncthreads();
    for (int w = 128; w; w >>= 1) {
        if (threadIdx.x < w) red[threadIdx.x] += red[threadIdx.x + w];
        __syncthreads();
    }
    if (threadIdx.x == 0)
        out[0] = 0.5f * red[0] / (float)((long long)BB * SS * SS);
}

// ---------------------------------------------------------------------------
extern "C" void kernel_launch(void* const* d_in, const int* in_sizes, int n_in,
                              void* d_out, int out_size, void* d_ws, size_t ws_size,
                              hipStream_t stream) {
    (void)in_sizes; (void)n_in; (void)out_size; (void)ws_size;
    const float* x = (const float*)d_in[0];
    float* out = (float*)d_out;

    float*    partials = (float*)d_ws;                          // 2048 floats
    _Float16* lnrm     = (_Float16*)((char*)d_ws + 16384);      // 16 MB f16

    norm_kernel<<<(BB * SS) / 8, 256, 0, stream>>>(x, lnrm);
    disp_kernel<<<BB * (SS / M_TILE), 128, 0, stream>>>(lnrm, partials);
    fin_kernel<<<1, 256, 0, stream>>>(partials, out);
}
